// AttentionEdgeReadout_32298154066171
// MI455X (gfx1250) — matrix operand order repaired and checked by reading, not compiled, not run on hardware
//
#include <hip/hip_runtime.h>
#include <cmath>

// ---------------------------------------------------------------------------
// Types for CDNA5 WMMA (wave32): V_WMMA_F32_16X16X32_BF16
// ---------------------------------------------------------------------------
typedef __attribute__((ext_vector_type(16))) __bf16 v16bf;
typedef __attribute__((ext_vector_type(8)))  float  v8f;

struct alignas(16) Frag32B { uint4 lo, hi; };

static __device__ inline v16bf make_frag(uint4 lo, uint4 hi) {
    Frag32B f{lo, hi};
    return __builtin_bit_cast(v16bf, f);
}

// low 32 bits of a generic pointer into __shared__ = LDS byte offset
static __device__ inline unsigned lds_off(const void* p) {
    return (unsigned)(size_t)p;
}

// ---------------------------------------------------------------------------
// f32 -> bf16 cast (grid-stride)
// ---------------------------------------------------------------------------
__global__ void cast_f32_bf16_kernel(const float* __restrict__ in,
                                     __bf16* __restrict__ out, long long n) {
    long long i = (long long)blockIdx.x * blockDim.x + threadIdx.x;
    long long stride = (long long)gridDim.x * blockDim.x;
    for (; i < n; i += stride) out[i] = (__bf16)in[i];
}

// ---------------------------------------------------------------------------
// Batched bf16 WMMA GEMM:  C[b] = A[b] (MxK) @ B[b] (KxN)  (B^T when TRANSB)
// Block tile 64x128, BK=32, 256 threads = 8 wave32 waves. Waves: 4 (m) x 2 (n),
// each wave owns a 16x64 strip = four 16x16 WMMA accumulators, so one A
// fragment feeds 4 WMMAs per staged tile. Tiles are staged with CDNA5
// GLOBAL_LOAD_ASYNC_TO_LDS_B128 (ASYNCcnt) where the memory layout allows.
// All problem dims here are multiples of 64/128/32, so no bounds checks.
// ---------------------------------------------------------------------------
#define BM 64
#define BN 128
#define BK 32
#define LDSS 40  // padded row stride in halves (80B = 5x16B, keeps 16B chunks aligned)

template <int TRANSB, int OUTBF16, int LEAKY>
__global__ __launch_bounds__(256)
void gemm_bf16_wmma_kernel(const __bf16* __restrict__ A,
                           const __bf16* __restrict__ Bm,
                           void* __restrict__ C,
                           int M, int N, int K,
                           long long sA, long long sB, long long sC,
                           const float* __restrict__ bias) {
    __shared__ __bf16 As[BM * LDSS];   // As[m][k]
    __shared__ __bf16 Bs[BN * LDSS];   // Bs[n][k] (n-major: B-frag reads contiguous)

    const int t    = threadIdx.x;
    const int lane = t & 31;
    const int wave = t >> 5;
    const int bz   = blockIdx.z;
    const int m0   = blockIdx.y * BM;
    const int n0   = blockIdx.x * BN;

    const __bf16* Ab = A  + (long long)bz * sA;
    const __bf16* Bb = Bm + (long long)bz * sB;

    const int wm = wave & 3;    // 0..3 -> 16-row sub-tile
    const int wn = wave >> 2;   // 0..1 -> 64-col sub-tile
    const int l  = lane & 15;
    const int hi = lane >> 4;

    v8f acc[4] = {v8f{}, v8f{}, v8f{}, v8f{}};

    // cooperative-load decomposition
    const int arow = t >> 2, akc = t & 3;   // A: 64 rows x 4 chunks of 8 halves (16B)
    const int brow = t >> 1, bkc = t & 1;   // B^T: 128 n-rows x 2 chunks of 16 halves (32B)
    const int bk   = t >> 3, bnc = t & 7;   // B row-major: 32 k-rows x 8 chunks of 16 n

    // per-thread LDS destinations (constant across K loop)
    const unsigned aLds = lds_off(&As[arow * LDSS + akc * 8]);
    const unsigned bLds = lds_off(&Bs[brow * LDSS + bkc * 16]);

    for (int k0 = 0; k0 < K; k0 += BK) {
        // ---- stage A tile: async copy straight into LDS (ASYNCcnt) ----
        {
            const __bf16* g = Ab + (long long)(m0 + arow) * K + k0 + akc * 8;
            asm volatile("global_load_async_to_lds_b128 %0, %1, off"
                         :: "v"(aLds), "v"(g) : "memory");
        }
        // ---- stage B tile into Bs[n][k] ----
        if (TRANSB) {
            // Bm is [N_total x K] row-major: 32B per n-row chunk, contiguous in k.
            // INST_OFFSET applies to both LDS and global sides (ISA 10.x), so the
            // second 16B chunk reuses the same base addresses with offset:16.
            const __bf16* g = Bb + (long long)(n0 + brow) * K + k0 + bkc * 16;
            asm volatile("global_load_async_to_lds_b128 %0, %1, off"
                         :: "v"(bLds), "v"(g) : "memory");
            asm volatile("global_load_async_to_lds_b128 %0, %1, off offset:16"
                         :: "v"(bLds), "v"(g) : "memory");
        } else {
            // Bm is [K x N] row-major; transpose on the way into LDS.
            const __bf16* g = Bb + (long long)(k0 + bk) * N + n0 + bnc * 16;
            uint4 bv0 = *(const uint4*)(g);
            uint4 bv1 = *(const uint4*)(g + 8);
            const __bf16* pv0 = (const __bf16*)&bv0;
            const __bf16* pv1 = (const __bf16*)&bv1;
#pragma unroll
            for (int j = 0; j < 8; ++j) {
                Bs[(bnc * 16 + j) * LDSS + bk]     = pv0[j];
                Bs[(bnc * 16 + 8 + j) * LDSS + bk] = pv1[j];
            }
        }
        // speculative prefetch of the next A tile (gfx1250 global_prefetch)
        __builtin_prefetch((const void*)(Ab + (long long)(m0 + arow) * K + (k0 + BK) + akc * 8), 0, 1);

        // all async LDS writes of this wave must land before the barrier
        asm volatile("s_wait_asynccnt 0x0" ::: "memory");
        __syncthreads();

        // ---- A fragment (ISA 16-bit A 16x32 layout):
        // lanes 0-15: K {0..7, 16..23}; lanes 16-31: K {8..15, 24..31}
        const __bf16* ap = &As[(wm * 16 + l) * LDSS];
        v16bf afrag = make_frag(*(const uint4*)(ap + hi * 8),
                                *(const uint4*)(ap + 16 + hi * 8));

        // ---- four B fragments: lane n = l, K {hi*8..+7, 16+hi*8..+7} (matches A) ----
#pragma unroll
        for (int tt = 0; tt < 4; ++tt) {
            const __bf16* bp = &Bs[(wn * 64 + tt * 16 + l) * LDSS];
            v16bf bfrag = make_frag(*(const uint4*)(bp + hi * 8), *(const uint4*)(bp + 16 + hi * 8));
            acc[tt] = __builtin_amdgcn_wmma_f32_16x16x32_bf16(
                false, afrag, false, bfrag, (short)0, acc[tt], false, false);
        }
        __syncthreads();
    }

    // ---- store C (ISA f32 C/D layout: M = vgpr + 8*hi, N = lane&15) ----
    const int gmBase = m0 + wm * 16 + hi * 8;
#pragma unroll
    for (int tt = 0; tt < 4; ++tt) {
        union { v8f v; float f[8]; } u;
        u.v = acc[tt];
        const int gn = n0 + wn * 64 + tt * 16 + l;
        const float bv = LEAKY ? bias[gn] : 0.0f;
#pragma unroll
        for (int v = 0; v < 8; ++v) {
            float x = u.f[v] + bv;
            if (LEAKY) x = (x >= 0.0f) ? x : 0.1f * x;
            long long idx = (long long)bz * sC + (long long)(gmBase + v) * N + gn;
            if (OUTBF16) ((__bf16*)C)[idx] = (__bf16)x;
            else         ((float*)C)[idx]  = x;
        }
    }
}

// ---------------------------------------------------------------------------
// Row softmax (scale applied here): P[row] = softmax(S[row] * scale), bf16 out
// ---------------------------------------------------------------------------
__global__ __launch_bounds__(256)
void softmax_rows_kernel(const float* __restrict__ S, __bf16* __restrict__ P,
                         int ncols, float scale) {
    __shared__ float red[256];
    const long long row = blockIdx.x;
    const float* srow = S + row * ncols;
    __bf16* prow = P + row * ncols;
    const int t = threadIdx.x;

    float m = -INFINITY;
    for (int c = t; c < ncols; c += 256) m = fmaxf(m, srow[c] * scale);
    red[t] = m; __syncthreads();
    for (int s = 128; s > 0; s >>= 1) {
        if (t < s) red[t] = fmaxf(red[t], red[t + s]);
        __syncthreads();
    }
    m = red[0]; __syncthreads();

    float sum = 0.0f;
    for (int c = t; c < ncols; c += 256) sum += __expf(srow[c] * scale - m);
    red[t] = sum; __syncthreads();
    for (int s = 128; s > 0; s >>= 1) {
        if (t < s) red[t] += red[t + s];
        __syncthreads();
    }
    const float inv = 1.0f / red[0];

    for (int c = t; c < ncols; c += 256)
        prow[c] = (__bf16)(__expf(srow[c] * scale - m) * inv);
}

// ---------------------------------------------------------------------------
// thresh[r] = H[r,:] . W2 + b2   (H already has bias + LeakyReLU applied)
// ---------------------------------------------------------------------------
__global__ void thresh_kernel(const float* __restrict__ H, const float* __restrict__ W2,
                              const float* __restrict__ b2, float* __restrict__ thr,
                              int rows, int cf) {
    int r = blockIdx.x * blockDim.x + threadIdx.x;
    if (r >= rows) return;
    const float* hr = H + (long long)r * cf;
    float acc = b2[0];
    for (int j = 0; j < cf; ++j) acc += hr[j] * W2[j];
    thr[r] = acc;
}

// ---------------------------------------------------------------------------
// A[row] = clip(relu(S[row]*scale - thr[row]) / (rowsum + 1e-9), 0, 1)
// ---------------------------------------------------------------------------
__global__ __launch_bounds__(256)
void finalize_kernel(const float* __restrict__ S, const float* __restrict__ thr,
                     float* __restrict__ Aout, int ncols, float scale) {
    __shared__ float buf[2048];
    __shared__ float red[256];
    const long long row = blockIdx.x;
    const float th = thr[row];
    const float* srow = S + row * ncols;
    float* orow = Aout + row * ncols;
    const int t = threadIdx.x;

    float sum = 0.0f;
    for (int c = t; c < ncols; c += 256) {
        float v = fmaxf(srow[c] * scale - th, 0.0f);
        buf[c] = v;
        sum += v;
    }
    red[t] = sum; __syncthreads();
    for (int s = 128; s > 0; s >>= 1) {
        if (t < s) red[t] += red[t + s];
        __syncthreads();
    }
    const float inv = 1.0f / (red[0] + 1e-9f);
    for (int c = t; c < ncols; c += 256)
        orow[c] = fminf(buf[c] * inv, 1.0f);
}

// ---------------------------------------------------------------------------
// Host orchestration
// ---------------------------------------------------------------------------
template <int TRANSB, int OUTBF16, int LEAKY>
static void launch_gemm(const __bf16* A, const __bf16* Bm, void* C,
                        int M, int N, int K,
                        long long sA, long long sB, long long sC, int batch,
                        const float* bias, hipStream_t stream) {
    dim3 grid(N / BN, M / BM, batch), block(256);
    gemm_bf16_wmma_kernel<TRANSB, OUTBF16, LEAKY><<<grid, block, 0, stream>>>(
        A, Bm, C, M, N, K, sA, sB, sC, bias);
}

extern "C" void kernel_launch(void* const* d_in, const int* in_sizes, int n_in,
                              void* d_out, int out_size, void* d_ws, size_t ws_size,
                              hipStream_t stream) {
    (void)in_sizes; (void)n_in; (void)out_size; (void)ws_size;
    constexpr int Bn = 8, Nn = 2048, Dd = 512, CF = 128;
    constexpr long long ND = (long long)Nn * Dd;   // per-batch stride of [N,D]
    constexpr long long NN = (long long)Nn * Nn;   // per-batch stride of [N,N]
    constexpr int MF = Bn * Nn;                    // 16384 flat rows
    const float scale = 0.044194173824159216f;     // 1/sqrt(512)

    const float* X   = (const float*)d_in[0];
    const float* Wq1 = (const float*)d_in[1];
    const float* Wk1 = (const float*)d_in[2];
    const float* Wv1 = (const float*)d_in[3];
    const float* Wo1 = (const float*)d_in[4];
    const float* Wq2 = (const float*)d_in[5];
    const float* Wk2 = (const float*)d_in[6];
    const float* W1  = (const float*)d_in[7];
    const float* b1  = (const float*)d_in[8];
    const float* W2  = (const float*)d_in[9];
    const float* b2  = (const float*)d_in[10];
    float* Aout = (float*)d_out;

    // workspace carve-up (256B aligned)
    char* ws = (char*)d_ws;
    size_t off = 0;
    auto alloc = [&](size_t bytes) -> char* {
        char* p = ws + off;
        off = (off + bytes + 255) & ~(size_t)255;
        return p;
    };
    __bf16* Xb   = (__bf16*)alloc((size_t)MF * Dd * 2);
    __bf16* Wq1b = (__bf16*)alloc((size_t)Dd * Dd * 2);
    __bf16* Wk1b = (__bf16*)alloc((size_t)Dd * Dd * 2);
    __bf16* Wv1b = (__bf16*)alloc((size_t)Dd * Dd * 2);
    __bf16* Wo1b = (__bf16*)alloc((size_t)Dd * Dd * 2);
    __bf16* Wq2b = (__bf16*)alloc((size_t)Dd * Dd * 2);
    __bf16* Wk2b = (__bf16*)alloc((size_t)Dd * Dd * 2);
    __bf16* W1b  = (__bf16*)alloc((size_t)Dd * CF * 2);
    __bf16* q1b  = (__bf16*)alloc((size_t)MF * Dd * 2);
    __bf16* k1b  = (__bf16*)alloc((size_t)MF * Dd * 2);
    __bf16* v1b  = (__bf16*)alloc((size_t)MF * Dd * 2);
    float*  Sf   = (float*) alloc((size_t)Bn * NN * 4);   // s1, later reused as s2
    __bf16* p1b  = (__bf16*)alloc((size_t)Bn * NN * 2);
    float*  Hf   = (float*) alloc((size_t)MF * CF * 4);
    float*  thr  = (float*) alloc((size_t)MF * 4);
    // buffer aliasing (lifetimes are disjoint):
    __bf16* att0b = q1b;   // free after s1
    __bf16* attb  = k1b;   // free after s1
    __bf16* q2b   = v1b;   // free after p1@v1
    __bf16* k2b   = Xb;    // free after v1 projection

    auto cast = [&](const float* src, __bf16* dst, long long n) {
        cast_f32_bf16_kernel<<<dim3(2048), dim3(256), 0, stream>>>(src, dst, n);
    };

    // 1) casts to bf16
    cast(X,   Xb,   (long long)MF * Dd);
    cast(Wq1, Wq1b, (long long)Dd * Dd);
    cast(Wk1, Wk1b, (long long)Dd * Dd);
    cast(Wv1, Wv1b, (long long)Dd * Dd);
    cast(Wo1, Wo1b, (long long)Dd * Dd);
    cast(Wq2, Wq2b, (long long)Dd * Dd);
    cast(Wk2, Wk2b, (long long)Dd * Dd);
    cast(W1,  W1b,  (long long)Dd * CF);

    // 2) projections q1/k1/v1 = X @ W  (flat 16384 x 512 x 512, bf16 out)
    launch_gemm<0, 1, 0>(Xb, Wq1b, q1b, MF, Dd, Dd, 0, 0, 0, 1, nullptr, stream);
    launch_gemm<0, 1, 0>(Xb, Wk1b, k1b, MF, Dd, Dd, 0, 0, 0, 1, nullptr, stream);
    launch_gemm<0, 1, 0>(Xb, Wv1b, v1b, MF, Dd, Dd, 0, 0, 0, 1, nullptr, stream);

    // 3) s1 = q1 @ k1^T  (batched 8x [2048 x 2048 x 512], f32 out)
    launch_gemm<1, 0, 0>(q1b, k1b, Sf, Nn, Nn, Dd, ND, ND, NN, Bn, nullptr, stream);

    // 4) p1 = softmax(s1 * scale)  -> bf16
    softmax_rows_kernel<<<dim3(MF), dim3(256), 0, stream>>>(Sf, p1b, Nn, scale);

    // 5) att0 = p1 @ v1  (batched 8x [2048 x 512 x 2048], bf16 out)
    launch_gemm<0, 1, 0>(p1b, v1b, att0b, Nn, Dd, Nn, NN, ND, ND, Bn, nullptr, stream);

    // 6) att = att0 @ Wo1  (flat, bf16 out)
    launch_gemm<0, 1, 0>(att0b, Wo1b, attb, MF, Dd, Dd, 0, 0, 0, 1, nullptr, stream);

    // 7) q2 / k2 = att @ Wq2 / Wk2  (bf16 out)
    launch_gemm<0, 1, 0>(attb, Wq2b, q2b, MF, Dd, Dd, 0, 0, 0, 1, nullptr, stream);
    launch_gemm<0, 1, 0>(attb, Wk2b, k2b, MF, Dd, Dd, 0, 0, 0, 1, nullptr, stream);

    // 8) h = LeakyReLU(att @ W1 + b1)  (flat 16384 x 128 x 512, f32 out)
    launch_gemm<0, 0, 1>(attb, W1b, Hf, MF, CF, Dd, 0, 0, 0, 1, b1, stream);

    // 9) thresh = h @ W2 + b2
    thresh_kernel<<<dim3((MF + 255) / 256), dim3(256), 0, stream>>>(Hf, W2, b2, thr, MF, CF);

    // 10) s2 = q2 @ k2^T  (reuses Sf, f32 out)
    launch_gemm<1, 0, 0>(q2b, k2b, Sf, Nn, Nn, Dd, ND, ND, NN, Bn, nullptr, stream);

    // 11) A = clip(relu(s2*scale - thresh) / rowsum, 0, 1)
    finalize_kernel<<<dim3(MF), dim3(256), 0, stream>>>(Sf, thr, Aout, Nn, scale);
}
